// DFlashAttentionSlide_56925496541547
// MI455X (gfx1250) — compile-verified, matrix operand order checked by reading, not run on hardware
//
#include <hip/hip_runtime.h>

#define DEV static __device__ __forceinline__

typedef __attribute__((ext_vector_type(16))) _Float16 v16h;
typedef __attribute__((ext_vector_type(8)))  float    v8f;

constexpr int H = 32, HKV = 8, D = 128, HALF = 64;
constexpr int L = 128, T = 1024, S = 4096, HID = 4096;
constexpr int REP = H / HKV;
constexpr float SCALE = 0.088388347648318447f; // 128^-0.5

// ---- WMMA fragment loaders (CDNA5 16-bit A/B layout, wave32) ----
// 16-bit A-matrix 16x32: lanes 0-15 hold rows M=0..15 with K = {0..7,16..23};
// lanes 16-31 hold the same rows with K = {8..15,24..31}. B mirrors per-lane
// with lane = column N. Element e -> k = (e&7) + 16*(e>>3) + 8*laneHi.

DEV v16h load_frag_rows(const float* __restrict__ base, int ld, int rowBase,
                        int kBase, int lane) {
  int lid = lane & 15, hi = (lane >> 4) & 1;
  const float* p = base + (size_t)(rowBase + lid) * ld + kBase + hi * 8;
  v16h r;
#pragma unroll
  for (int e = 0; e < 8; ++e) r[e] = (_Float16)p[e];
#pragma unroll
  for (int e = 0; e < 8; ++e) r[8 + e] = (_Float16)p[16 + e];
  return r;
}

DEV v16h load_frag_cols(const float* __restrict__ base, int ld, int kBase,
                        int nBase, int lane) {
  int lid = lane & 15, hi = (lane >> 4) & 1;
  v16h r;
#pragma unroll
  for (int e = 0; e < 8; ++e)
    r[e] = (_Float16)base[(size_t)(kBase + hi * 8 + e) * ld + nBase + lid];
#pragma unroll
  for (int e = 0; e < 8; ++e)
    r[8 + e] = (_Float16)base[(size_t)(kBase + 16 + hi * 8 + e) * ld + nBase + lid];
  return r;
}

DEV v8f wmma16x16x32(v16h a, v16h b, v8f c) {
  return __builtin_amdgcn_wmma_f32_16x16x32_f16(false, a, false, b, (short)0, c,
                                                false, false);
}

// ---- C = A @ B^T  (A: [M,K], B: [N,K], C: [M,N]) ----
// One wave computes a 32x64 tile: 2 A-frags x 4 B-frags -> 8 WMMAs per k-step.
__global__ __launch_bounds__(32) void gemm_xwt(const float* __restrict__ A,
                                               const float* __restrict__ B,
                                               float* __restrict__ C,
                                               int N, int K) {
  int lane = threadIdx.x;
  int lid = lane & 15, hi = (lane >> 4) & 1;
  int nB = blockIdx.x * 64;
  int mB = blockIdx.y * 32;
  v8f vz = {};
  v8f acc[2][4];
#pragma unroll
  for (int i = 0; i < 2; ++i)
#pragma unroll
    for (int j = 0; j < 4; ++j) acc[i][j] = vz;

  for (int k = 0; k < K; k += 32) {
    if (k + 64 <= K) {
      __builtin_prefetch(A + (size_t)(mB + lid) * K + k + 32, 0, 3);
      __builtin_prefetch(A + (size_t)(mB + 16 + lid) * K + k + 32, 0, 3);
#pragma unroll
      for (int j = 0; j < 4; ++j)
        __builtin_prefetch(B + (size_t)(nB + j * 16 + lid) * K + k + 32, 0, 3);
    }
    v16h a0 = load_frag_rows(A, K, mB, k, lane);
    v16h a1 = load_frag_rows(A, K, mB + 16, k, lane);
#pragma unroll
    for (int j = 0; j < 4; ++j) {
      v16h b = load_frag_rows(B, K, nB + j * 16, k, lane);
      acc[0][j] = wmma16x16x32(a0, b, acc[0][j]);
      acc[1][j] = wmma16x16x32(a1, b, acc[1][j]);
    }
  }
#pragma unroll
  for (int i = 0; i < 2; ++i)
#pragma unroll
    for (int j = 0; j < 4; ++j)
#pragma unroll
      for (int r = 0; r < 8; ++r)
        C[(size_t)(mB + i * 16 + r + 8 * hi) * N + nB + j * 16 + lid] =
            acc[i][j][r];
}

// ---- per-(token,head) mean-subtract LayerNorm + RoPE, scatter to dst ----
__global__ __launch_bounds__(128) void norm_rope(const float* __restrict__ src,
                                                 float* __restrict__ dst,
                                                 const float* __restrict__ w,
                                                 const float* __restrict__ cs,
                                                 const float* __restrict__ sn,
                                                 int nHead, int srcRowStride,
                                                 size_t dstHeadStride,
                                                 int dstTokOff) {
  int vec = blockIdx.x;
  int t = vec / nHead, h = vec - t * nHead;
  int d = threadIdx.x;
  float x = src[(size_t)t * srcRowStride + h * D + d];
  __shared__ float red[128];
  __shared__ float ybuf[128];
  red[d] = x;
  __syncthreads();
#pragma unroll
  for (int s = 64; s > 0; s >>= 1) {
    if (d < s) red[d] += red[d + s];
    __syncthreads();
  }
  float mu = red[0] * (1.0f / D);
  __syncthreads();
  float xc = x - mu;
  red[d] = xc * xc;
  __syncthreads();
#pragma unroll
  for (int s = 64; s > 0; s >>= 1) {
    if (d < s) red[d] += red[d + s];
    __syncthreads();
  }
  float var = red[0] * (1.0f / D);
  float y = xc * rsqrtf(var + 1e-6f) * w[d];
  ybuf[d] = y;
  __syncthreads();
  float rot = (d < HALF) ? -ybuf[d + HALF] : ybuf[d - HALF];
  float c = cs[(size_t)t * D + d];
  float s2 = sn[(size_t)t * D + d];
  dst[h * dstHeadStride + (size_t)(t + dstTokOff) * D + d] = y * c + rot * s2;
}

// ---- K/V stream assembly: cache tails + new V tokens ----
__global__ void assemble_streams(const float* __restrict__ cache,
                                 const float* __restrict__ vraw,
                                 float* __restrict__ kst,
                                 float* __restrict__ vst) {
  size_t i = (size_t)blockIdx.x * blockDim.x + threadIdx.x;
  constexpr size_t TAIL = (size_t)HKV * (S - T) * D;
  constexpr size_t VNEW = (size_t)HKV * T * D;
  if (i < TAIL) {
    int d = (int)(i % D);
    size_t q = i / D;
    int t = (int)(q % (S - T));
    int h = (int)(q / (S - T));
    size_t dsti = ((size_t)h * S + t) * D + d;
    kst[dsti] = cache[((size_t)h * S + (T + t)) * D + d];
    vst[dsti] = cache[((size_t)(HKV + h) * S + (T + t)) * D + d];
  } else if (i < TAIL + VNEW) {
    size_t j = i - TAIL;
    int d = (int)(j % D);
    size_t q = j / D;
    int t = (int)(q % T);
    int h = (int)(q / T);
    vst[((size_t)h * S + (S - T) + t) * D + d] =
        vraw[(size_t)t * (HKV * D) + h * D + d];
  }
}

// ---- flash attention: one wave per (head, 16-row q tile), online softmax ----
__global__ __launch_bounds__(32) void attn_fwd(const float* __restrict__ qf,
                                               const float* __restrict__ kst,
                                               const float* __restrict__ vst,
                                               float* __restrict__ out) {
  int h = blockIdx.x;
  int qt = blockIdx.y;
  int lane = threadIdx.x;
  int lid = lane & 15, hi = (lane >> 4) & 1;
  int kvh = h / REP;
  const float* Q = qf + ((size_t)h * L + qt * 16) * D;
  const float* Kh = kst + (size_t)kvh * S * D;
  const float* Vh = vst + (size_t)kvh * S * D;

  v16h qa[4];
#pragma unroll
  for (int d = 0; d < 4; ++d) qa[d] = load_frag_rows(Q, D, 0, d * 32, lane);

  float rowmax[8], rowsum[8];
  v8f vzero = {};
  v8f acc[8];
#pragma unroll
  for (int r = 0; r < 8; ++r) { rowmax[r] = -1e30f; rowsum[r] = 0.f; }
#pragma unroll
  for (int dt = 0; dt < 8; ++dt) acc[dt] = vzero;

  __shared__ float Pb[16][33];  // P staging, padded to dodge bank conflicts

  for (int sc = 0; sc < S; sc += 32) {
    v8f c0 = vzero, c1 = vzero;
#pragma unroll
    for (int d = 0; d < 4; ++d) {
      v16h b0 = load_frag_rows(Kh, D, sc, d * 32, lane);
      v16h b1 = load_frag_rows(Kh, D, sc + 16, d * 32, lane);
      c0 = wmma16x16x32(qa[d], b0, c0);
      c1 = wmma16x16x32(qa[d], b1, c1);
    }
    int j0 = sc + lid, j1 = sc + 16 + lid;
    float s0[8], s1[8], mnew[8], alpha[8];
#pragma unroll
    for (int r = 0; r < 8; ++r) {
      int lim = (S - L) + qt * 16 + r + 8 * hi;  // causal: j <= S-L+m
      s0[r] = c0[r] * SCALE + (j0 <= lim ? 0.f : -1e30f);
      s1[r] = c1[r] * SCALE + (j1 <= lim ? 0.f : -1e30f);
      float t = fmaxf(s0[r], s1[r]);
#pragma unroll
      for (int o = 1; o < 16; o <<= 1) t = fmaxf(t, __shfl_xor(t, o, 32));
      mnew[r] = fmaxf(rowmax[r], t);
      alpha[r] = __expf(rowmax[r] - mnew[r]);
      rowmax[r] = mnew[r];
    }
    float p0[8], p1[8];
#pragma unroll
    for (int r = 0; r < 8; ++r) {
      p0[r] = __expf(s0[r] - mnew[r]);
      p1[r] = __expf(s1[r] - mnew[r]);
      float ps = p0[r] + p1[r];
#pragma unroll
      for (int o = 1; o < 16; o <<= 1) ps += __shfl_xor(ps, o, 32);
      rowsum[r] = rowsum[r] * alpha[r] + ps;
    }
#pragma unroll
    for (int dt = 0; dt < 8; ++dt)
#pragma unroll
      for (int r = 0; r < 8; ++r) acc[dt][r] *= alpha[r];

    // C-layout -> A-layout for P via LDS
#pragma unroll
    for (int r = 0; r < 8; ++r) {
      Pb[r + 8 * hi][lid] = p0[r];
      Pb[r + 8 * hi][16 + lid] = p1[r];
    }
    __syncthreads();
    v16h pa;
#pragma unroll
    for (int e = 0; e < 8; ++e) pa[e] = (_Float16)Pb[lid][e + 8 * hi];
#pragma unroll
    for (int e = 0; e < 8; ++e) pa[8 + e] = (_Float16)Pb[lid][16 + e + 8 * hi];
    __syncthreads();

#pragma unroll
    for (int dt = 0; dt < 8; ++dt) {
      v16h bv = load_frag_cols(Vh, D, sc, dt * 16, lane);
      acc[dt] = wmma16x16x32(pa, bv, acc[dt]);
    }
  }

#pragma unroll
  for (int dt = 0; dt < 8; ++dt)
#pragma unroll
    for (int r = 0; r < 8; ++r) {
      int m = qt * 16 + r + 8 * hi;
      out[(size_t)m * (H * D) + h * D + dt * 16 + lid] = acc[dt][r] / rowsum[r];
    }
}

extern "C" void kernel_launch(void* const* d_in, const int* in_sizes, int n_in,
                              void* d_out, int out_size, void* d_ws,
                              size_t ws_size, hipStream_t stream) {
  const float* x = (const float*)d_in[0];
  const float* x_ctx = (const float*)d_in[1];
  const float* cos_q = (const float*)d_in[2];
  const float* sin_q = (const float*)d_in[3];
  const float* cos_k = (const float*)d_in[4];
  const float* sin_k = (const float*)d_in[5];
  const float* cache = (const float*)d_in[6];
  // d_in[7] = causal_mask (unused; mask computed analytically)
  const float* Wq = (const float*)d_in[8];
  const float* Wk = (const float*)d_in[9];
  const float* Wv = (const float*)d_in[10];
  const float* Wo = (const float*)d_in[11];
  const float* qw = (const float*)d_in[12];
  const float* kw = (const float*)d_in[13];
  float* out = (float*)d_out;

  float* ws = (float*)d_ws;
  float* c = ws;      ws += (size_t)T * HID;
  float* q_raw = ws;  ws += (size_t)L * H * D;
  float* k_raw = ws;  ws += (size_t)T * HKV * D;
  float* v_raw = ws;  ws += (size_t)T * HKV * D;
  float* qf = ws;     ws += (size_t)H * L * D;
  float* kst = ws;    ws += (size_t)HKV * S * D;
  float* vst = ws;    ws += (size_t)HKV * S * D;
  float* attn_o = ws; ws += (size_t)L * H * D;

  // c = concat(x_ctx, x)
  hipMemcpyAsync(c, x_ctx, sizeof(float) * (size_t)(T - L) * HID,
                 hipMemcpyDeviceToDevice, stream);
  hipMemcpyAsync(c + (size_t)(T - L) * HID, x, sizeof(float) * (size_t)L * HID,
                 hipMemcpyDeviceToDevice, stream);

  // projections (q only needs the last L tokens, i.e. x itself)
  gemm_xwt<<<dim3((H * D) / 64, L / 32), 32, 0, stream>>>(x, Wq, q_raw, H * D, HID);
  gemm_xwt<<<dim3((HKV * D) / 64, T / 32), 32, 0, stream>>>(c, Wk, k_raw, HKV * D, HID);
  gemm_xwt<<<dim3((HKV * D) / 64, T / 32), 32, 0, stream>>>(c, Wv, v_raw, HKV * D, HID);

  // norm + rope; k written directly into stream tail region
  norm_rope<<<L * H, D, 0, stream>>>(q_raw, qf, qw, cos_q, sin_q, H, H * D,
                                     (size_t)L * D, 0);
  norm_rope<<<T * HKV, D, 0, stream>>>(k_raw, kst, kw, cos_k, sin_k, HKV,
                                       HKV * D, (size_t)S * D, S - T);

  size_t total = (size_t)HKV * (S - T) * D + (size_t)HKV * T * D;
  assemble_streams<<<(unsigned)((total + 255) / 256), 256, 0, stream>>>(
      cache, v_raw, kst, vst);

  attn_fwd<<<dim3(H, L / 16), 32, 0, stream>>>(qf, kst, vst, attn_o);

  gemm_xwt<<<dim3(HID / 64, L / 32), 32, 0, stream>>>(attn_o, Wo, out, HID, HID);
}
